// CvtSelfAttention_35553739276472
// MI455X (gfx1250) — matrix operand order repaired and checked by reading, not compiled, not run on hardware
//
#include <hip/hip_runtime.h>
#include <math.h>

// ---------------------------------------------------------------------------
// CvT self-attention for MI455X (gfx1250, wave32, WMMA).
// Pipeline: dwconv+BN -> f16, WMMA projection GEMMs, WMMA attention.
// Round 2: double-buffered fragment loads in the accumulating WMMA loops so
// global-load latency overlaps the matrix pipe (prior asm showed
// load -> s_wait_loadcnt 0 -> wmma lockstep in the GEMM/PV loops).
// ---------------------------------------------------------------------------

typedef _Float16 v16h __attribute__((ext_vector_type(16)));
typedef _Float16 v8h  __attribute__((ext_vector_type(8)));
typedef float    v8f  __attribute__((ext_vector_type(8)));

union V16 { v16h v; v8h h[2]; };

constexpr int   BATCH = 32;
constexpr int   LQ    = 1025;   // 1 cls + 32*32
constexpr int   LK    = 257;    // 1 cls + 16*16
constexpr int   EMB   = 768;
constexpr int   NH    = 12;
constexpr int   HD    = 64;
constexpr int   LQP   = 1040;   // LQ padded to 16
constexpr int   LKP   = 288;    // LK padded to 32 (9 K-steps of 32)
constexpr int   MQ    = BATCH * LQ;  // 32800 (divisible by 32)
constexpr int   MKV   = BATCH * LK;  // 8224  (divisible by 32)
constexpr float EPSF  = 1e-5f;
constexpr float SCALE = 0.03608439182435161f; // 768^-0.5

static __device__ __forceinline__ v8f wmma_f16(v16h a, v16h b, v8f c) {
  return __builtin_amdgcn_wmma_f32_16x16x32_f16(false, a, false, b, (short)0, c,
                                                false, false);
}

// ------------------------- weight f32 -> f16 -------------------------------
__global__ void cvt_w_kernel(const float* __restrict__ src,
                             _Float16* __restrict__ dst, int n) {
  int i = blockIdx.x * blockDim.x + threadIdx.x;
  if (i < n) dst[i] = (_Float16)src[i];
}

// --------------- depthwise 3x3 conv + folded BN -> f16 tokens --------------
// out: (B, LC, E) token-major, row 0 = cls token copied from x.
template <int STRIDE>
__global__ void dwconv_bn_kernel(const float* __restrict__ x,
                                 const float* __restrict__ cw,
                                 const float* __restrict__ gam,
                                 const float* __restrict__ bet,
                                 const float* __restrict__ mu,
                                 const float* __restrict__ var,
                                 _Float16* __restrict__ out) {
  constexpr int OH = 32 / STRIDE;
  constexpr int LC = 1 + OH * OH;
  int id = blockIdx.x * blockDim.x + threadIdx.x;
  if (id >= BATCH * LC * EMB) return;
  int c = id % EMB;
  int t = (id / EMB) % LC;
  int b = id / (EMB * LC);
  const float* xb = x + (size_t)b * LQ * EMB;
  if (t == 0) {  // cls token passes through
    out[((size_t)b * LC) * EMB + c] = (_Float16)xb[c];
    return;
  }
  int oy = (t - 1) / OH, ox = (t - 1) % OH;
  float acc = 0.f;
#pragma unroll
  for (int dy = 0; dy < 3; ++dy) {
    int iy = oy * STRIDE + dy - 1;
    if (iy < 0 || iy >= 32) continue;
#pragma unroll
    for (int dx = 0; dx < 3; ++dx) {
      int ix = ox * STRIDE + dx - 1;
      if (ix < 0 || ix >= 32) continue;
      acc += cw[c * 9 + dy * 3 + dx] * xb[(size_t)(1 + iy * 32 + ix) * EMB + c];
    }
  }
  float sc = gam[c] * rsqrtf(var[c] + EPSF);
  float sb = bet[c] - mu[c] * sc;
  out[((size_t)b * LC + t) * EMB + c] = (_Float16)(acc * sc + sb);
}

// --------------------- projection GEMM: P = A @ W^T + b --------------------
// A: (Mtot, 768) f16 row-major.  W16: (768, 768) f16 row-major (n, k).
// One wave computes a 16x64 tile; block = 8 waves = 32(M) x 256(N).
// Double-buffered K-loop: fragments for step kk+1 are issued before the
// WMMAs of step kk so vmem latency hides under the matrix pipe.
// MODE 0: Q -> qbuf[(b*12+h)*1040 + l][d]  (scaled by SCALE)
// MODE 1: K -> kbuf[(b*12+h)*288  + t][d]
// MODE 2: V -> vT  [(b*12+h)*64   + d][t]  (transposed)
template <int MODE>
__global__ void __launch_bounds__(256)
gemm_proj_kernel(const _Float16* __restrict__ A,
                 const _Float16* __restrict__ W16,
                 const float* __restrict__ bias,
                 _Float16* __restrict__ dst) {
  constexpr int LTOK  = (MODE == 0) ? LQ : LK;
  constexpr int KSTEP = EMB / 32;  // 24
  int lane = threadIdx.x & 31;
  int wv   = threadIdx.x >> 5;
  int m0   = blockIdx.x * 32 + (wv & 1) * 16;
  int n0   = blockIdx.y * 256 + (wv >> 1) * 64;
  int l15  = lane & 15;
  int hiA  = (lane < 16) ? 0 : 8;   // A frag K split per ISA layout
  int hiB  = (lane < 16) ? 0 : 8;   // B frag K split per ISA layout

  const _Float16* arow = A + (size_t)(m0 + l15) * EMB + hiA;
  const _Float16* brow = W16 + (size_t)(n0 + l15) * EMB + hiB;

  v8f acc[4] = {};
  V16 a_cur, b_cur[4];
  a_cur.h[0] = *(const v8h*)(arow);
  a_cur.h[1] = *(const v8h*)(arow + 16);
#pragma unroll
  for (int nt = 0; nt < 4; ++nt) {
    const _Float16* bp = brow + (size_t)nt * 16 * EMB;
    b_cur[nt].h[0] = *(const v8h*)bp;
    b_cur[nt].h[1] = *(const v8h*)(bp + 16);
  }

#pragma unroll 4
  for (int kk = 0; kk < KSTEP; ++kk) {
    V16 a_nxt, b_nxt[4];
    if (kk + 1 < KSTEP) {
      int k1 = (kk + 1) * 32;
      a_nxt.h[0] = *(const v8h*)(arow + k1);
      a_nxt.h[1] = *(const v8h*)(arow + k1 + 16);
#pragma unroll
      for (int nt = 0; nt < 4; ++nt) {
        const _Float16* bp = brow + (size_t)nt * 16 * EMB + k1;
        b_nxt[nt].h[0] = *(const v8h*)bp;
        b_nxt[nt].h[1] = *(const v8h*)(bp + 16);
      }
    }
#pragma unroll
    for (int nt = 0; nt < 4; ++nt)
      acc[nt] = wmma_f16(a_cur.v, b_cur[nt].v, acc[nt]);
    if (kk + 1 < KSTEP) {
      a_cur = a_nxt;
#pragma unroll
      for (int nt = 0; nt < 4; ++nt) b_cur[nt] = b_nxt[nt];
    }
  }

  int mrow = (lane < 16) ? 0 : 8;
#pragma unroll
  for (int nt = 0; nt < 4; ++nt) {
    int n = n0 + nt * 16 + l15;
    float bv = bias[n];
    int h = n >> 6, d = n & 63;
#pragma unroll
    for (int r = 0; r < 8; ++r) {
      int m = m0 + r + mrow;
      int b = m / LTOK;
      int t = m - b * LTOK;
      float val = acc[nt][r] + bv;
      if (MODE == 0) {
        dst[((size_t)(b * NH + h) * LQP + t) * HD + d] = (_Float16)(val * SCALE);
      } else if (MODE == 1) {
        dst[((size_t)(b * NH + h) * LKP + t) * HD + d] = (_Float16)val;
      } else {
        dst[((size_t)(b * NH + h) * HD + d) * LKP + t] = (_Float16)val;
      }
    }
  }
}

// ------------------------------- attention ---------------------------------
// One wave handles 16 query rows of one (b,h): 36 WMMAs for scores,
// f32 softmax, probs -> LDS (C-layout -> A-layout), 36 WMMAs for P@V.
constexpr int MTILES = (LQ + 15) / 16;  // 65

__global__ void __launch_bounds__(128)
attn_kernel(const _Float16* __restrict__ Q, const _Float16* __restrict__ K,
            const _Float16* __restrict__ VT, float* __restrict__ out) {
  __shared__ _Float16 Slds[4][16][LKP];  // 36 KB: per-wave prob tiles
  int lane  = threadIdx.x & 31;
  int wv    = threadIdx.x >> 5;
  int mtile = blockIdx.y * 4 + wv;
  if (mtile >= MTILES) return;
  int bh  = blockIdx.x;  // 0..383 = b*12+h
  int m0  = mtile * 16;
  int l15 = lane & 15;
  int hiA = (lane < 16) ? 0 : 8;
  int hiB = (lane < 16) ? 0 : 8;
  int mrw = (lane < 16) ? 0 : 8;

  // Q fragments for this 16x64 strip (K-dim = head dim, two steps of 32).
  V16 aq[2];
  const _Float16* qp = Q + ((size_t)bh * LQP + m0 + l15) * HD;
#pragma unroll
  for (int s = 0; s < 2; ++s) {
    aq[s].h[0] = *(const v8h*)(qp + s * 32 + hiA);
    aq[s].h[1] = *(const v8h*)(qp + s * 32 + hiA + 16);
  }

  // Scores: S[t] is a 16x16 f32 tile over key columns t*16..t*16+15.
  // (Independent accumulators: compiler already pipelines these loads.)
  v8f S[LKP / 16];
#pragma unroll
  for (int t = 0; t < LKP / 16; ++t) {
    v8f acc = {};
    const _Float16* kp = K + ((size_t)bh * LKP + t * 16 + l15) * HD + hiB;
    V16 b0, b1;
    b0.h[0] = *(const v8h*)kp;
    b0.h[1] = *(const v8h*)(kp + 16);
    b1.h[0] = *(const v8h*)(kp + 32);
    b1.h[1] = *(const v8h*)(kp + 48);
    acc  = wmma_f16(aq[0].v, b0.v, acc);
    acc  = wmma_f16(aq[1].v, b1.v, acc);
    S[t] = acc;
  }

  // Mask padded key columns (col >= 257) by overwrite (NaN-proof).
#pragma unroll
  for (int t = 16; t < LKP / 16; ++t) {
    if (t * 16 + l15 >= LK) {
#pragma unroll
      for (int r = 0; r < 8; ++r) S[t][r] = -__builtin_inff();
    }
  }

  // Row softmax: rows live across the 16-lane halves of the wave.
  float rinv[8];
#pragma unroll
  for (int r = 0; r < 8; ++r) {
    float mx = -__builtin_inff();
#pragma unroll
    for (int t = 0; t < LKP / 16; ++t) mx = fmaxf(mx, S[t][r]);
#pragma unroll
    for (int off = 8; off >= 1; off >>= 1) mx = fmaxf(mx, __shfl_xor(mx, off, 32));
    float sum = 0.f;
#pragma unroll
    for (int t = 0; t < LKP / 16; ++t) {
      float e = __expf(S[t][r] - mx);
      S[t][r] = e;
      sum += e;
    }
#pragma unroll
    for (int off = 8; off >= 1; off >>= 1) sum += __shfl_xor(sum, off, 32);
    rinv[r] = 1.0f / sum;
  }

  // C-layout -> A-layout via LDS (per-wave region, no barrier needed).
  _Float16* sl = &Slds[wv][0][0];
#pragma unroll
  for (int t = 0; t < LKP / 16; ++t) {
#pragma unroll
    for (int r = 0; r < 8; ++r) {
      sl[(r + mrw) * LKP + t * 16 + l15] = (_Float16)S[t][r];
    }
  }
  asm volatile("s_wait_dscnt 0" ::: "memory");

  // ctx = P @ V : K-dim = key index (9 steps of 32), N = head dim (4 tiles).
  // Double-buffered V fragments: step s+1 loads issue before step s WMMAs.
  const _Float16* vbase = VT + ((size_t)bh * HD + l15) * LKP + hiB;
  v8f C4[4] = {};
  V16 vb_cur[4];
#pragma unroll
  for (int nt = 0; nt < 4; ++nt) {
    const _Float16* vp = vbase + (size_t)nt * 16 * LKP;
    vb_cur[nt].h[0] = *(const v8h*)vp;
    vb_cur[nt].h[1] = *(const v8h*)(vp + 16);
  }
#pragma unroll
  for (int s = 0; s < LKP / 32; ++s) {
    V16 vb_nxt[4];
    if (s + 1 < LKP / 32) {
      int k1 = (s + 1) * 32;
#pragma unroll
      for (int nt = 0; nt < 4; ++nt) {
        const _Float16* vp = vbase + (size_t)nt * 16 * LKP + k1;
        vb_nxt[nt].h[0] = *(const v8h*)vp;
        vb_nxt[nt].h[1] = *(const v8h*)(vp + 16);
      }
    }
    V16 a;
    const _Float16* pp = sl + (size_t)l15 * LKP + s * 32 + hiA;
    a.h[0] = *(const v8h*)pp;
    a.h[1] = *(const v8h*)(pp + 16);
#pragma unroll
    for (int nt = 0; nt < 4; ++nt)
      C4[nt] = wmma_f16(a.v, vb_cur[nt].v, C4[nt]);
    if (s + 1 < LKP / 32) {
#pragma unroll
      for (int nt = 0; nt < 4; ++nt) vb_cur[nt] = vb_nxt[nt];
    }
  }

  // Normalize rows and store fp32 output (B, LQ, 768).
  int b = bh / NH, h = bh - b * NH;
#pragma unroll
  for (int nt = 0; nt < 4; ++nt) {
    int d = h * HD + nt * 16 + l15;
#pragma unroll
    for (int r = 0; r < 8; ++r) {
      int ltok = m0 + r + mrw;
      if (ltok < LQ)
        out[((size_t)b * LQ + ltok) * EMB + d] = C4[nt][r] * rinv[r];
    }
  }
}

// ---------------------------------------------------------------------------
extern "C" void kernel_launch(void* const* d_in, const int* in_sizes, int n_in,
                              void* d_out, int out_size, void* d_ws,
                              size_t ws_size, hipStream_t stream) {
  (void)in_sizes; (void)n_in; (void)out_size; (void)ws_size;
  const float* x    = (const float*)d_in[0];
  // d_in[1]=height(32), d_in[2]=width(32) — fixed by the problem instance.
  const float* cw_q = (const float*)d_in[3];
  const float* g_q  = (const float*)d_in[4];
  const float* be_q = (const float*)d_in[5];
  const float* mu_q = (const float*)d_in[6];
  const float* vr_q = (const float*)d_in[7];
  const float* W_q  = (const float*)d_in[8];
  const float* bi_q = (const float*)d_in[9];
  const float* cw_k = (const float*)d_in[10];
  const float* g_k  = (const float*)d_in[11];
  const float* be_k = (const float*)d_in[12];
  const float* mu_k = (const float*)d_in[13];
  const float* vr_k = (const float*)d_in[14];
  const float* W_k  = (const float*)d_in[15];
  const float* bi_k = (const float*)d_in[16];
  const float* cw_v = (const float*)d_in[17];
  const float* g_v  = (const float*)d_in[18];
  const float* be_v = (const float*)d_in[19];
  const float* mu_v = (const float*)d_in[20];
  const float* vr_v = (const float*)d_in[21];
  const float* W_v  = (const float*)d_in[22];
  const float* bi_v = (const float*)d_in[23];

  char* p = (char*)d_ws;
  auto carve = [&](size_t halfs) {
    _Float16* r = (_Float16*)p;
    p += (halfs * 2 + 255) & ~(size_t)255;
    return r;
  };
  _Float16* qin  = carve((size_t)MQ * EMB);
  _Float16* kin  = carve((size_t)MKV * EMB);
  _Float16* vin  = carve((size_t)MKV * EMB);
  _Float16* W16q = carve((size_t)EMB * EMB);
  _Float16* W16k = carve((size_t)EMB * EMB);
  _Float16* W16v = carve((size_t)EMB * EMB);
  _Float16* qbuf = carve((size_t)BATCH * NH * LQP * HD);
  _Float16* kbuf = carve((size_t)BATCH * NH * LKP * HD);
  _Float16* vtb  = carve((size_t)BATCH * NH * HD * LKP);

  // Stage 0: weights to f16; zero vT so padded key cols multiply as 0*0.
  const int NW = EMB * EMB;
  hipMemsetAsync(vtb, 0, (size_t)BATCH * NH * HD * LKP * 2, stream);
  cvt_w_kernel<<<(NW + 255) / 256, 256, 0, stream>>>(W_q, W16q, NW);
  cvt_w_kernel<<<(NW + 255) / 256, 256, 0, stream>>>(W_k, W16k, NW);
  cvt_w_kernel<<<(NW + 255) / 256, 256, 0, stream>>>(W_v, W16v, NW);

  // Stage 1: depthwise conv + BN -> f16 token-major activations.
  dwconv_bn_kernel<1><<<(BATCH * LQ * EMB) / 256, 256, 0, stream>>>(
      x, cw_q, g_q, be_q, mu_q, vr_q, qin);
  dwconv_bn_kernel<2><<<(BATCH * LK * EMB) / 256, 256, 0, stream>>>(
      x, cw_k, g_k, be_k, mu_k, vr_k, kin);
  dwconv_bn_kernel<2><<<(BATCH * LK * EMB) / 256, 256, 0, stream>>>(
      x, cw_v, g_v, be_v, mu_v, vr_v, vin);

  // Stage 2: WMMA projection GEMMs into head-split layouts.
  gemm_proj_kernel<0><<<dim3(MQ / 32, EMB / 256), 256, 0, stream>>>(
      qin, W16q, bi_q, qbuf);
  gemm_proj_kernel<1><<<dim3(MKV / 32, EMB / 256), 256, 0, stream>>>(
      kin, W16k, bi_k, kbuf);
  gemm_proj_kernel<2><<<dim3(MKV / 32, EMB / 256), 256, 0, stream>>>(
      vin, W16v, bi_v, vtb);

  // Stage 3: WMMA attention.
  attn_kernel<<<dim3(BATCH * NH, (MTILES + 3) / 4), 128, 0, stream>>>(
      qbuf, kbuf, vtb, (float*)d_out);
}
